// CrystalGCNVAE_16260746183181
// MI455X (gfx1250) — matrix operand order repaired and checked by reading, not compiled, not run on hardware
//
#include <hip/hip_runtime.h>
#include <hip/hip_bf16.h>

// ---------------- problem constants (match reference) ----------------
constexpr int N  = 50000;   // total nodes (multiple of 16)
constexpr int P  = 5000;    // nodes per graph
constexpr int G  = 10;      // graphs
constexpr int E  = 800000;  // edges
constexpr int F  = 92;      // node feature dim
constexpr int H  = 4;       // heads
constexpr int D1 = 32, D2 = 64, D3 = 128;
constexpr int L  = 64;      // latent
constexpr int HID = 64;     // hidden
constexpr float SLOPE = 0.2f;

typedef __attribute__((ext_vector_type(16))) _Float16 v16h;
typedef __attribute__((ext_vector_type(8)))  _Float16 v8h;
typedef __attribute__((ext_vector_type(4)))  _Float16 v4h;
typedef __attribute__((ext_vector_type(8)))  float    v8f;

// ---------------- helpers ----------------
__device__ __forceinline__ float atomicMaxFloat(float* addr, float value) {
    if (value >= 0.0f)
        return __int_as_float(atomicMax((int*)addr, __float_as_int(value)));
    else
        return __uint_as_float(atomicMin((unsigned int*)addr, __float_as_uint(value)));
}

__global__ void k_fill(float* __restrict__ p, float v, int n) {
    int i = blockIdx.x * 256 + threadIdx.x;
    if (i < n) p[i] = v;
}

// zero-pad copy: src[srcRows, srcCols] -> dst[*, dstCols]
__global__ void k_pad(const float* __restrict__ src, float* __restrict__ dst,
                      int srcRows, int srcCols, int dstCols, int total) {
    int idx = blockIdx.x * 256 + threadIdx.x;
    if (idx >= total) return;
    int r = idx / dstCols, c = idx % dstCols;
    dst[idx] = (r < srcRows && c < srcCols) ? src[(size_t)r * srcCols + c] : 0.0f;
}

// ---------------- WMMA GEMM: C[N,Mout] = act(A[N,K] @ W[K,Mw] (+bias)) ----------------
// Requirements (guaranteed by caller via padding):
//   K % 32 == 0 (compile-time), N rows multiple of 16, gridDim.x * 64 == Mw.
// block = 128 threads = 4 waves; block tile 16 rows x 64 cols; wave tile 16x16.
// fp32 -> f16 staged through LDS, fp32 accumulate via v_wmma_f32_16x16x32_f16.
// ACT: 0 = none (no bias), 1 = +bias, 2 = relu(+bias)
template <int ACT, int K>
__global__ __launch_bounds__(128)
void gemm_wmma(const float* __restrict__ A, const float* __restrict__ W,
               const float* __restrict__ bias, float* __restrict__ C,
               int Mw, int Mout) {
    constexpr int AST = 40;              // padded LDS stride in halfs (80B: bank-friendly)
    __shared__ _Float16 sA[16 * AST];    // [row][k]   (k contiguous)
    __shared__ _Float16 sBT[64 * AST];   // [col][k]   (k contiguous, transposed)
    const int tid  = threadIdx.x;
    const int wave = tid >> 5;
    const int lane = tid & 31;
    const int row0 = blockIdx.y * 16;
    const int col0 = blockIdx.x * 64;
    const int hi   = lane >> 4;
    const int l15  = lane & 15;

    // A staging task: thread t loads 4 consecutive floats of row ar at k-offset ak0
    const int ar  = tid >> 3;            // 0..15
    const int ak0 = (tid & 7) << 2;      // 0,4,..,28

    v8f acc = {};
#pragma unroll
    for (int kk = 0; kk < K; kk += 32) {
        // ---- stage A (16x32): one aligned float4 per thread ----
        {
            const float4 fv = *(const float4*)(A + (size_t)(row0 + ar) * K + (kk + ak0));
            v4h hv = { (_Float16)fv.x, (_Float16)fv.y, (_Float16)fv.z, (_Float16)fv.w };
            *(v4h*)(&sA[ar * AST + ak0]) = hv;
        }
        // ---- stage B transposed (64 cols x 32 k): 4 tasks/thread,
        //      each task = 4 coalesced b32 loads down one column + one b64 LDS store
#pragma unroll
        for (int it = 0; it < 4; ++it) {
            int task = tid + it * 128;        // 0..511
            int c    = task & 63;             // column within tile
            int kb   = (task >> 6) << 2;      // 0,4,..,28
            const float* wp = W + (size_t)(kk + kb) * Mw + (col0 + c);
            v4h hv = { (_Float16)wp[0],
                       (_Float16)wp[(size_t)Mw],
                       (_Float16)wp[(size_t)2 * Mw],
                       (_Float16)wp[(size_t)3 * Mw] };
            *(v4h*)(&sBT[c * AST + kb]) = hv;
        }
        __syncthreads();

        // ---- fragments (ISA wave32 16-bit layouts): two b128 LDS loads each ----
        // A: row = l15, halves i<8 -> k = 8*hi + i ; i>=8 -> k = 16 + 8*hi + (i-8)
        v8h a0 = *(const v8h*)(&sA[l15 * AST + (hi << 3)]);
        v8h a1 = *(const v8h*)(&sA[l15 * AST + 16 + (hi << 3)]);
        v16h a = __builtin_shufflevector(a0, a1, 0, 1, 2, 3, 4, 5, 6, 7,
                                                 8, 9, 10, 11, 12, 13, 14, 15);
        // B: col = wave*16 + l15, halves i<8 -> k = 8*hi + i ; i>=8 -> k = 16 + 8*hi + (i-8)
        const int bcol = (wave << 4) + l15;
        v8h b0 = *(const v8h*)(&sBT[bcol * AST + (hi << 3)]);
        v8h b1 = *(const v8h*)(&sBT[bcol * AST + 16 + (hi << 3)]);
        v16h b = __builtin_shufflevector(b0, b1, 0, 1, 2, 3, 4, 5, 6, 7,
                                                 8, 9, 10, 11, 12, 13, 14, 15);

        acc = __builtin_amdgcn_wmma_f32_16x16x32_f16(
            /*neg_a=*/false, a, /*neg_b=*/false, b,
            /*c_mod=*/(short)0, acc, /*reuse_a=*/false, /*reuse_b=*/false);
        __syncthreads();
    }

    // ---- writeback: col = lane&15, row = r + 8*hi ----
    const int ccol = col0 + (wave << 4) + l15;
    if (ccol < Mout) {
        float bv = 0.0f;
        if (ACT >= 1) bv = bias[ccol];
#pragma unroll
        for (int r = 0; r < 8; ++r) {
            int crow = row0 + r + (hi << 3);
            float v = acc[r] + bv;
            if (ACT == 2) v = fmaxf(v, 0.0f);
            C[(size_t)crow * Mout + ccol] = v;
        }
    }
}

// ---------------- GAT edge machinery ----------------
__global__ void k_elr(const float* __restrict__ h, const float* __restrict__ al,
                      const float* __restrict__ ar, float* __restrict__ el,
                      float* __restrict__ er, int D) {
    int idx = blockIdx.x * 256 + threadIdx.x;
    if (idx >= N * H) return;
    int n = idx >> 2, hh = idx & 3;
    int HD = H * D;
    const float* hp  = h + (size_t)n * HD + hh * D;
    const float* alp = al + hh * D;
    const float* arp = ar + hh * D;
    float sl = 0.0f, sr = 0.0f;
    for (int d = 0; d < D; ++d) { float v = hp[d]; sl += v * alp[d]; sr += v * arp[d]; }
    el[idx] = sl; er[idx] = sr;
}

__global__ void k_edge_max(const int* __restrict__ dsts, const int* __restrict__ offs,
                           const float* __restrict__ el, const float* __restrict__ er,
                           float* __restrict__ m) {
    int idx = blockIdx.x * 256 + threadIdx.x;
    if (idx >= E * H) return;
    int e = idx >> 2, hh = idx & 3;
    int dst = dsts[e];
    int src = (dst / P) * P + offs[e];
    float s = el[src * 4 + hh] + er[dst * 4 + hh];
    s = (s > 0.0f) ? s : SLOPE * s;
    atomicMaxFloat(&m[dst * 4 + hh], s);
}

__global__ void k_edge_sum(const int* __restrict__ dsts, const int* __restrict__ offs,
                           const float* __restrict__ el, const float* __restrict__ er,
                           const float* __restrict__ m, float* __restrict__ ex,
                           float* __restrict__ ssum) {
    int idx = blockIdx.x * 256 + threadIdx.x;
    if (idx >= E * H) return;
    int e = idx >> 2, hh = idx & 3;
    int dst = dsts[e];
    int src = (dst / P) * P + offs[e];
    float s = el[src * 4 + hh] + er[dst * 4 + hh];
    s = (s > 0.0f) ? s : SLOPE * s;
    float x = expf(s - m[dst * 4 + hh]);
    ex[idx] = x;
    atomicAdd(&ssum[dst * 4 + hh], x);
}

__global__ void k_edge_scatter(const int* __restrict__ dsts, const int* __restrict__ offs,
                               const float* __restrict__ ex, const float* __restrict__ ssum,
                               const float* __restrict__ hpre, float* __restrict__ agg, int D) {
    int idx = blockIdx.x * 256 + threadIdx.x;
    if (idx >= E * H) return;
    int e = idx >> 2, hh = idx & 3;
    int dst = dsts[e];
    int src = (dst / P) * P + offs[e];
    float alpha = ex[idx] / ssum[dst * 4 + hh];
    int HD = 4 * D;
    const float* hs = hpre + (size_t)src * HD + hh * D;
    float* ag = agg + (size_t)dst * HD + hh * D;
    for (int d = 0; d < D; ++d) atomicAdd(&ag[d], alpha * hs[d]);
}

__global__ void k_bias_elu(float* __restrict__ x, const float* __restrict__ b,
                           int HD, int total) {
    int idx = blockIdx.x * 256 + threadIdx.x;
    if (idx >= total) return;
    int c = idx % HD;
    float v = x[idx] + b[c];
    x[idx] = (v > 0.0f) ? v : expm1f(v);
}

// ---------------- global attention pooling ----------------
__global__ void k_gate(const float* __restrict__ gh, const float* __restrict__ gW2,
                       const float* __restrict__ gb2, float* __restrict__ gate) {
    int n = blockIdx.x * 256 + threadIdx.x;
    if (n >= N) return;
    float acc = gb2[0];
    const float* gp = gh + (size_t)n * HID;
    for (int j = 0; j < HID; ++j) acc += gp[j] * gW2[j];
    gate[n] = acc;
}

__global__ void k_pool_max(const float* __restrict__ gate, float* __restrict__ gm) {
    int n = blockIdx.x * 256 + threadIdx.x;
    if (n >= N) return;
    atomicMaxFloat(&gm[n / P], gate[n]);
}

__global__ void k_pool_sum(const float* __restrict__ gate, const float* __restrict__ gm,
                           float* __restrict__ gex, float* __restrict__ gs) {
    int n = blockIdx.x * 256 + threadIdx.x;
    if (n >= N) return;
    float x = expf(gate[n] - gm[n / P]);
    gex[n] = x;
    atomicAdd(&gs[n / P], x);
}

__global__ void k_pool_emb(const float* __restrict__ h3, const float* __restrict__ gex,
                           const float* __restrict__ gs, float* __restrict__ emb) {
    int g = blockIdx.x;
    int c = blockIdx.y * 128 + threadIdx.x;  // 0..511
    float inv = 1.0f / gs[g];
    const float* hp = h3 + (size_t)g * P * 512;
    const float* gp = gex + g * P;
    float acc = 0.0f;
    for (int n = 0; n < P; ++n) acc += gp[n] * hp[(size_t)n * 512 + c];
    emb[g * 512 + c] = acc * inv;
}

// ---------------- VAE head + decoder front ----------------
__global__ void k_vae(const float* __restrict__ emb, const float* __restrict__ Wmu,
                      const float* __restrict__ bmu, const float* __restrict__ Wlv,
                      const float* __restrict__ blv, const float* __restrict__ eps,
                      float* __restrict__ out_mu, float* __restrict__ out_lv,
                      float* __restrict__ z) {
    int idx = blockIdx.x * 64 + threadIdx.x;
    if (idx >= G * L) return;
    int g = idx >> 6, l = idx & 63;
    const float* eg = emb + g * 512;
    float mu = bmu[l], lv = blv[l];
    for (int k = 0; k < 512; ++k) {
        float v = eg[k];
        mu += v * Wmu[k * 64 + l];
        lv += v * Wlv[k * 64 + l];
    }
    out_mu[idx] = mu;
    out_lv[idx] = lv;
    z[idx] = mu + expf(0.5f * lv) * eps[idx];
}

__global__ void k_decp(const float* __restrict__ z, const float* __restrict__ Wp,
                       const float* __restrict__ bp, float* __restrict__ hz) {
    int idx = blockIdx.x * 256 + threadIdx.x;
    if (idx >= G * 2 * HID) return;
    int g = idx >> 7, j = idx & 127;
    const float* zg = z + g * L;
    float acc = bp[j];
    for (int k = 0; k < L; ++k) acc += zg[k] * Wp[k * 128 + j];
    hz[idx] = fmaxf(acc, 0.0f);
}

__global__ void k_bcast(const float* __restrict__ hz, float* __restrict__ hexp) {
    int idx = blockIdx.x * 256 + threadIdx.x;
    if (idx >= N * 128) return;
    int n = idx >> 7, j = idx & 127;
    hexp[idx] = hz[(n / P) * 128 + j];
}

// ---------------- host orchestration ----------------
static inline int nblk(int n, int b = 256) { return (n + b - 1) / b; }

extern "C" void kernel_launch(void* const* d_in, const int* in_sizes, int n_in,
                              void* d_out, int out_size, void* d_ws, size_t ws_size,
                              hipStream_t stream) {
    (void)in_sizes; (void)n_in; (void)out_size; (void)ws_size;
    const float* node_feat = (const float*)d_in[0];
    const float* eps  = (const float*)d_in[1];
    const int*   edst = (const int*)d_in[2];
    const int*   eoff = (const int*)d_in[3];
    const float* W1 = (const float*)d_in[4];   const float* al1 = (const float*)d_in[5];
    const float* ar1 = (const float*)d_in[6];  const float* b1 = (const float*)d_in[7];
    const float* W2 = (const float*)d_in[8];   const float* al2 = (const float*)d_in[9];
    const float* ar2 = (const float*)d_in[10]; const float* b2 = (const float*)d_in[11];
    const float* W3 = (const float*)d_in[12];  const float* al3 = (const float*)d_in[13];
    const float* ar3 = (const float*)d_in[14]; const float* b3 = (const float*)d_in[15];
    const float* gW1 = (const float*)d_in[16]; const float* gb1 = (const float*)d_in[17];
    const float* gW2 = (const float*)d_in[18]; const float* gb2 = (const float*)d_in[19];
    const float* Wmu = (const float*)d_in[20]; const float* bmu = (const float*)d_in[21];
    const float* Wlv = (const float*)d_in[22]; const float* blv = (const float*)d_in[23];
    const float* Wp  = (const float*)d_in[24]; const float* bp  = (const float*)d_in[25];
    const float* Wd1 = (const float*)d_in[26]; const float* bd1 = (const float*)d_in[27];
    const float* Wd2 = (const float*)d_in[28]; const float* bd2 = (const float*)d_in[29];

    // workspace layout (256B aligned blocks)
    char* ws = (char*)d_ws;
    size_t off = 0;
    auto alloc = [&](size_t bytes) -> float* {
        float* p = (float*)(ws + off);
        off += (bytes + 255) & ~(size_t)255;
        return p;
    };
    float* bufA   = alloc((size_t)N * 512 * 4);   // GEMM outputs / h_pre
    float* bufB   = alloc((size_t)N * 512 * 4);   // aggregated activations / inputs
    float* Apad   = alloc((size_t)N * 96 * 4);    // node_feat padded to K=96
    float* W1pad  = alloc((size_t)96 * 128 * 4);  // W1 padded rows 92->96
    float* Wd2pad = alloc((size_t)64 * 128 * 4);  // Wd2 padded cols 92->128
    float* el     = alloc((size_t)N * H * 4);
    float* er     = alloc((size_t)N * H * 4);
    float* mbuf   = alloc((size_t)N * H * 4);
    float* sbuf   = alloc((size_t)N * H * 4);
    float* exbuf  = alloc((size_t)E * H * 4);
    float* emb    = alloc((size_t)G * 512 * 4);
    float* zbuf   = alloc((size_t)G * L * 4);
    float* hzbuf  = alloc((size_t)G * 2 * HID * 4);
    float* gm     = alloc((size_t)G * 4);
    float* gs     = alloc((size_t)G * 4);

    const float NEG_INF = -__builtin_inff();

    // -------- padding passes (make all GEMMs guard-free) --------
    k_pad<<<nblk(N * 96), 256, 0, stream>>>(node_feat, Apad, N, F, 96, N * 96);
    k_pad<<<nblk(96 * 128), 256, 0, stream>>>(W1, W1pad, F, 128, 128, 96 * 128);
    k_pad<<<nblk(64 * 128), 256, 0, stream>>>(Wd2, Wd2pad, 64, F, 128, 64 * 128);

    // -------- 3 GAT layers --------
    auto edge_phase = [&](const float* al, const float* ar, const float* b, int D) {
        const int M = H * D;
        k_elr<<<nblk(N * H), 256, 0, stream>>>(bufA, al, ar, el, er, D);
        k_fill<<<nblk(N * H), 256, 0, stream>>>(mbuf, NEG_INF, N * H);
        k_fill<<<nblk(N * H), 256, 0, stream>>>(sbuf, 0.0f, N * H);
        k_fill<<<nblk(N * M), 256, 0, stream>>>(bufB, 0.0f, N * M);
        k_edge_max<<<nblk(E * H), 256, 0, stream>>>(edst, eoff, el, er, mbuf);
        k_edge_sum<<<nblk(E * H), 256, 0, stream>>>(edst, eoff, el, er, mbuf, exbuf, sbuf);
        k_edge_scatter<<<nblk(E * H), 256, 0, stream>>>(edst, eoff, exbuf, sbuf, bufA, bufB, D);
        k_bias_elu<<<nblk(N * M), 256, 0, stream>>>(bufB, b, M, N * M);
    };

    // layer 1: Apad[N,96] @ W1pad[96,128]
    gemm_wmma<0, 96><<<dim3(128 / 64, N / 16), 128, 0, stream>>>(Apad, W1pad, nullptr, bufA, 128, 128);
    edge_phase(al1, ar1, b1, D1);
    // layer 2: bufB[N,128] @ W2[128,256]
    gemm_wmma<0, 128><<<dim3(256 / 64, N / 16), 128, 0, stream>>>(bufB, W2, nullptr, bufA, 256, 256);
    edge_phase(al2, ar2, b2, D2);
    // layer 3: bufB[N,256] @ W3[256,512]
    gemm_wmma<0, 256><<<dim3(512 / 64, N / 16), 128, 0, stream>>>(bufB, W3, nullptr, bufA, 512, 512);
    edge_phase(al3, ar3, b3, D3);
    // bufB now holds h3 [N, 512]

    // -------- global attention pooling --------
    gemm_wmma<2, 512><<<dim3(1, N / 16), 128, 0, stream>>>(bufB, gW1, gb1, bufA, 64, 64);
    float* gate = el;  // reuse
    float* gex  = er;
    k_gate<<<nblk(N), 256, 0, stream>>>(bufA, gW2, gb2, gate);
    k_fill<<<1, 32, 0, stream>>>(gm, NEG_INF, G);
    k_fill<<<1, 32, 0, stream>>>(gs, 0.0f, G);
    k_pool_max<<<nblk(N), 256, 0, stream>>>(gate, gm);
    k_pool_sum<<<nblk(N), 256, 0, stream>>>(gate, gm, gex, gs);
    k_pool_emb<<<dim3(G, 4), 128, 0, stream>>>(bufB, gex, gs, emb);

    // -------- VAE head (mu, logvar, z) --------
    float* out    = (float*)d_out;
    float* out_mu = out + (size_t)N * F;
    float* out_lv = out_mu + (size_t)G * L;
    k_vae<<<(G * L + 63) / 64, 64, 0, stream>>>(emb, Wmu, bmu, Wlv, blv, eps,
                                                out_mu, out_lv, zbuf);

    // -------- decoder --------
    k_decp<<<nblk(G * 2 * HID), 256, 0, stream>>>(zbuf, Wp, bp, hzbuf);
    k_bcast<<<nblk(N * 128), 256, 0, stream>>>(hzbuf, bufA);  // h_exp [N,128]
    gemm_wmma<2, 128><<<dim3(1, N / 16), 128, 0, stream>>>(bufA, Wd1, bd1, bufB, 64, 64);
    gemm_wmma<1, 64><<<dim3(2, N / 16), 128, 0, stream>>>(bufB, Wd2pad, bd2, out, 128, F);
}
